// GraphAttentionNetwork_14224931684477
// MI455X (gfx1250) — matrix operand order repaired and checked by reading, not compiled, not run on hardware
//
#include <hip/hip_runtime.h>
#include <hip/hip_bf16.h>

// ---------------------------------------------------------------------------
// GATv2 x3 + pooling + head for MI455X (gfx1250, wave32, WMMA).
// Dense transforms: v_wmma_f32_16x16x32_f16, B pre-packed in fragment order,
// A loaded with 128-bit vector loads, 4 M-tiles per wave for B reuse.
// Edge softmax/aggregation via float atomics into L2 (node buffers fit L2).
// ---------------------------------------------------------------------------

typedef __attribute__((ext_vector_type(16))) _Float16 v16h;
typedef __attribute__((ext_vector_type(8)))  _Float16 v8h;
typedef __attribute__((ext_vector_type(8)))  float    v8f;

#define N_NODES 65536
#define N_GRAPH 128
#define NPG     512
#define N_EDGE  1048576
#define FDIM    128

// ---------------- utility kernels ----------------

__global__ void fill_f32_k(float* __restrict__ p, float v, int n) {
    int i = blockIdx.x * blockDim.x + threadIdx.x;
    if (i < n) p[i] = v;
}

// x [N x 8] f32 -> A16 [N x 32] f16 zero-padded in K
__global__ void pad_x_k(const float* __restrict__ x, _Float16* __restrict__ out, int n) {
    int idx = blockIdx.x * blockDim.x + threadIdx.x;
    if (idx >= n) return;                 // n = N*32
    int k = idx & 31, i = idx >> 5;
    out[idx] = (k < 8) ? (_Float16)x[i * 8 + k] : (_Float16)0.0f;
}

// generic f32 -> f16 convert
__global__ void cvt_f16_k(const float* __restrict__ in, _Float16* __restrict__ out, int n) {
    int i = blockIdx.x * blockDim.x + threadIdx.x;
    if (i < n) out[i] = (_Float16)in[i];
}

// Pack weight W [kin x Nc] f32 into WMMA-B fragment order, f16, K padded
// to kpad. Layout: for (ntile, ktile): 32 lanes x 16 contiguous halfs.
//   B[k][n], k = ktile*32 + (lane>>4)*8 + (i&7) + (i>>3)*16, n = ntile*16 + (lane&15)
__global__ void pack_b_k(const float* __restrict__ w, _Float16* __restrict__ bp,
                         int kin, int kpad, int Nc, int n /* = (Nc/16)*(kpad/32)*512 */) {
    int idx = blockIdx.x * blockDim.x + threadIdx.x;
    if (idx >= n) return;
    int i    = idx & 15;
    int lane = (idx >> 4) & 31;
    int t    = idx >> 9;                  // ntile*(kpad/32) + ktile
    int kt   = kpad >> 5;
    int ktile = t % kt;
    int ntile = t / kt;
    int k   = ktile * 32 + (lane >> 4) * 8 + (i & 7) + (i >> 3) * 16;
    int col = ntile * 16 + (lane & 15);
    bp[idx] = (k < kin) ? (_Float16)w[(size_t)k * Nc + col] : (_Float16)0.0f;
}

// ---------------- WMMA GEMM: C[M x Nc] = A[M x K] * B + bias --------------
// One wave computes a 64x16 strip (4 stacked 16x16 tiles, shared B fragment).
// K multiple of 32, M multiple of 64, Nc multiple of 16.
__global__ void gemm_wmma_k(const _Float16* __restrict__ A, int K,
                            const _Float16* __restrict__ Bp,   // packed fragments
                            const float* __restrict__ bias,
                            float* __restrict__ C, int M, int Nc) {
    int gtid = blockIdx.x * blockDim.x + threadIdx.x;
    int wave = gtid >> 5;
    int lane = gtid & 31;
    int tilesN = Nc >> 4;
    int mb    = wave / tilesN;
    int ntile = wave % tilesN;
    int m0 = mb << 6;
    if (m0 >= M) return;

    int hs = lane >> 4;          // lane[4]
    int r  = lane & 15;
    int kt = K >> 5;

    v8f acc[4] = {};
    const v16h* bp = (const v16h*)Bp + (size_t)ntile * kt * 32 + lane;
    // A fragment: lane holds row; halves 0..7 -> K=kk+8*hs+i, 8..15 -> +16
    const _Float16* abase = A + (size_t)(m0 + r) * K + hs * 8;

    for (int ktile = 0; ktile < kt; ++ktile) {
        v16h b = bp[(size_t)ktile * 32];
        int kk = ktile * 32;
#pragma unroll
        for (int mt = 0; mt < 4; ++mt) {
            const _Float16* ar = abase + (size_t)mt * 16 * K + kk;
            union { struct { v8h lo, hi; } s; v16h v; } u;
            u.s.lo = *(const v8h*)(ar);
            u.s.hi = *(const v8h*)(ar + 16);
            acc[mt] = __builtin_amdgcn_wmma_f32_16x16x32_f16(
                false, u.v, false, b, (short)0, acc[mt], false, false);
        }
    }

    int n = ntile * 16 + r;
    float bv = bias ? bias[n] : 0.0f;
#pragma unroll
    for (int mt = 0; mt < 4; ++mt)
#pragma unroll
        for (int v = 0; v < 8; ++v) {
            int m = m0 + mt * 16 + v + hs * 8;
            C[(size_t)m * Nc + n] = acc[mt][v] + bv;
        }
}

// ---------------- edge passes ----------------

__device__ __forceinline__ void atomicMaxF(float* addr, float val) {
    if (val >= 0.0f) atomicMax((int*)addr, __float_as_int(val));
    else             atomicMin((unsigned int*)addr, __float_as_uint(val));
}

template<int H, int C>
__global__ void edge_logits_max_k(const float* __restrict__ xl, const float* __restrict__ xr,
                                  const long long* __restrict__ src,
                                  const long long* __restrict__ dst,
                                  const float* __restrict__ att,
                                  float* __restrict__ logits,
                                  float* __restrict__ nodemax, int nE) {
    int idx = blockIdx.x * blockDim.x + threadIdx.x;
    if (idx >= nE * H) return;
    int e = idx / H, h = idx - e * H;
    int s = (int)src[e], d = (int)dst[e];
    const float* pl = xl + (size_t)s * FDIM + h * C;
    const float* pr = xr + (size_t)d * FDIM + h * C;
    float acc = 0.0f;
#pragma unroll
    for (int c = 0; c < C; ++c) {
        float v = pl[c] + pr[c];
        v = (v > 0.0f) ? v : 0.2f * v;      // leaky_relu 0.2
        acc += v * att[h * C + c];
    }
    logits[(size_t)e * H + h] = acc;
    atomicMaxF(nodemax + (size_t)d * H + h, acc);
}

template<int H, int C>
__global__ void edge_accum_k(const float* __restrict__ xl,
                             const long long* __restrict__ src,
                             const long long* __restrict__ dst,
                             const float* __restrict__ logits,
                             const float* __restrict__ nodemax,
                             float* __restrict__ den, float* __restrict__ agg, int nE) {
    int idx = blockIdx.x * blockDim.x + threadIdx.x;
    if (idx >= nE * H) return;
    int e = idx / H, h = idx - e * H;
    int s = (int)src[e], d = (int)dst[e];
    float m = nodemax[(size_t)d * H + h];
    if (!(m > -3.0e38f)) m = 0.0f;          // guard empty segments
    float a = __expf(logits[(size_t)e * H + h] - m);
    atomicAdd(den + (size_t)d * H + h, a);
    const float* pl = xl + (size_t)s * FDIM + h * C;
    float* pa = agg + (size_t)d * FDIM + h * C;
#pragma unroll
    for (int c = 0; c < C; ++c)
        atomicAdd(pa + c, a * pl[c]);
}

template<int H, int C, bool LEAKY>
__global__ void finalize_k(const float* __restrict__ agg, const float* __restrict__ den,
                           const float* __restrict__ bias, float* __restrict__ out, int nN) {
    int idx = blockIdx.x * blockDim.x + threadIdx.x;
    if (idx >= nN * FDIM) return;
    int i = idx >> 7, f = idx & (FDIM - 1);
    int h = f / C;
    float o = agg[idx] / (den[(size_t)i * H + h] + 1e-16f) + bias[f];
    if (LEAKY) o = (o > 0.0f) ? o : 0.01f * o;
    out[idx] = o;
}

// ---------------- pooling: min/max/mean/sum per graph, leaky, -> f16 -------
__global__ void pool_k(const float* __restrict__ h, _Float16* __restrict__ pooled16) {
    int b = blockIdx.x;          // graph
    int f = threadIdx.x;         // feature 0..127
    const float* p = h + (size_t)b * NPG * FDIM + f;
    float mn = 3.4e38f, mx = -3.4e38f, sm = 0.0f;
    for (int i = 0; i < NPG; ++i) {
        float v = p[(size_t)i * FDIM];
        mn = fminf(mn, v); mx = fmaxf(mx, v); sm += v;
    }
    _Float16* row = pooled16 + (size_t)b * 512;
    float vals[4] = { mn, mx, sm * (1.0f / NPG), sm };
#pragma unroll
    for (int k = 0; k < 4; ++k) {
        float v = vals[k];
        v = (v > 0.0f) ? v : 0.01f * v;
        row[k * 128 + f] = (_Float16)v;
    }
}

// ---------------- host driver ----------------

extern "C" void kernel_launch(void* const* d_in, const int* in_sizes, int n_in,
                              void* d_out, int out_size, void* d_ws, size_t ws_size,
                              hipStream_t stream) {
    const int N = N_NODES, E = N_EDGE;

    const float*     x    = (const float*)d_in[0];
    const long long* ei   = (const long long*)d_in[1];
    const long long* esrc = ei;
    const long long* edst = ei + E;
    const float *w1l = (const float*)d_in[3],  *b1l = (const float*)d_in[4];
    const float *w1r = (const float*)d_in[5],  *b1r = (const float*)d_in[6];
    const float *att1 = (const float*)d_in[7], *bias1 = (const float*)d_in[8];
    const float *w2l = (const float*)d_in[9],  *b2l = (const float*)d_in[10];
    const float *w2r = (const float*)d_in[11], *b2r = (const float*)d_in[12];
    const float *att2 = (const float*)d_in[13],*bias2 = (const float*)d_in[14];
    const float *w3l = (const float*)d_in[15], *b3l = (const float*)d_in[16];
    const float *w3r = (const float*)d_in[17], *b3r = (const float*)d_in[18];
    const float *att3 = (const float*)d_in[19],*bias3 = (const float*)d_in[20];
    const float *wh  = (const float*)d_in[21], *bh  = (const float*)d_in[22];
    float* out = (float*)d_out;

    // workspace layout
    size_t off = 0;
    auto alloc = [&](size_t bytes) {
        void* p = (char*)d_ws + off;
        off += (bytes + 255) & ~(size_t)255;
        return p;
    };
    _Float16* A16  = (_Float16*)alloc((size_t)N * FDIM * 2);   // GEMM input
    _Float16* BPa  = (_Float16*)alloc(512 * 128 * 2);          // packed B (left)
    _Float16* BPb  = (_Float16*)alloc(512 * 128 * 2);          // packed B (right)
    float* XL   = (float*)alloc((size_t)N * FDIM * 4);
    float* XR   = (float*)alloc((size_t)N * FDIM * 4);
    float* LOG  = (float*)alloc((size_t)E * 8 * 4);
    float* MAXB = (float*)alloc((size_t)N * 8 * 4);
    float* DEN  = (float*)alloc((size_t)N * 8 * 4);
    float* AGG  = (float*)alloc((size_t)N * FDIM * 4);
    float* HB   = (float*)alloc((size_t)N * FDIM * 4);
    _Float16* P16 = (_Float16*)alloc(128 * 512 * 2);

    const int TB = 256;
    auto blk = [&](long long n) { return (int)((n + TB - 1) / TB); };
    auto gemm_grid = [&](int M, int Nc) {
        long long waves = (long long)(M / 64) * (Nc / 16);
        return blk(waves * 32);
    };

    // ===== Layer 1 (in 8 -> pad 32, H=8, C=16) =====
    pad_x_k<<<blk((long long)N * 32), TB, 0, stream>>>(x, A16, N * 32);
    pack_b_k<<<blk(8 * 1 * 512), TB, 0, stream>>>(w1l, BPa, 8, 32, 128, 8 * 1 * 512);
    pack_b_k<<<blk(8 * 1 * 512), TB, 0, stream>>>(w1r, BPb, 8, 32, 128, 8 * 1 * 512);
    gemm_wmma_k<<<gemm_grid(N, 128), TB, 0, stream>>>(A16, 32, BPa, b1l, XL, N, 128);
    gemm_wmma_k<<<gemm_grid(N, 128), TB, 0, stream>>>(A16, 32, BPb, b1r, XR, N, 128);

    fill_f32_k<<<blk((long long)N * 8), TB, 0, stream>>>(MAXB, -3.4e38f, N * 8);
    fill_f32_k<<<blk((long long)N * 8), TB, 0, stream>>>(DEN, 0.0f, N * 8);
    fill_f32_k<<<blk((long long)N * FDIM), TB, 0, stream>>>(AGG, 0.0f, N * FDIM);
    edge_logits_max_k<8, 16><<<blk((long long)E * 8), TB, 0, stream>>>(XL, XR, esrc, edst, att1, LOG, MAXB, E);
    edge_accum_k<8, 16><<<blk((long long)E * 8), TB, 0, stream>>>(XL, esrc, edst, LOG, MAXB, DEN, AGG, E);
    finalize_k<8, 16, true><<<blk((long long)N * FDIM), TB, 0, stream>>>(AGG, DEN, bias1, HB, N);

    // ===== Layer 2 (128 -> 128, H=8, C=16) =====
    cvt_f16_k<<<blk((long long)N * FDIM), TB, 0, stream>>>(HB, A16, N * FDIM);
    pack_b_k<<<blk(8 * 4 * 512), TB, 0, stream>>>(w2l, BPa, 128, 128, 128, 8 * 4 * 512);
    pack_b_k<<<blk(8 * 4 * 512), TB, 0, stream>>>(w2r, BPb, 128, 128, 128, 8 * 4 * 512);
    gemm_wmma_k<<<gemm_grid(N, 128), TB, 0, stream>>>(A16, 128, BPa, b2l, XL, N, 128);
    gemm_wmma_k<<<gemm_grid(N, 128), TB, 0, stream>>>(A16, 128, BPb, b2r, XR, N, 128);

    fill_f32_k<<<blk((long long)N * 8), TB, 0, stream>>>(MAXB, -3.4e38f, N * 8);
    fill_f32_k<<<blk((long long)N * 8), TB, 0, stream>>>(DEN, 0.0f, N * 8);
    fill_f32_k<<<blk((long long)N * FDIM), TB, 0, stream>>>(AGG, 0.0f, N * FDIM);
    edge_logits_max_k<8, 16><<<blk((long long)E * 8), TB, 0, stream>>>(XL, XR, esrc, edst, att2, LOG, MAXB, E);
    edge_accum_k<8, 16><<<blk((long long)E * 8), TB, 0, stream>>>(XL, esrc, edst, LOG, MAXB, DEN, AGG, E);
    finalize_k<8, 16, true><<<blk((long long)N * FDIM), TB, 0, stream>>>(AGG, DEN, bias2, HB, N);

    // ===== Layer 3 (128 -> 128, H=1, C=128, no activation) =====
    cvt_f16_k<<<blk((long long)N * FDIM), TB, 0, stream>>>(HB, A16, N * FDIM);
    pack_b_k<<<blk(8 * 4 * 512), TB, 0, stream>>>(w3l, BPa, 128, 128, 128, 8 * 4 * 512);
    pack_b_k<<<blk(8 * 4 * 512), TB, 0, stream>>>(w3r, BPb, 128, 128, 128, 8 * 4 * 512);
    gemm_wmma_k<<<gemm_grid(N, 128), TB, 0, stream>>>(A16, 128, BPa, b3l, XL, N, 128);
    gemm_wmma_k<<<gemm_grid(N, 128), TB, 0, stream>>>(A16, 128, BPb, b3r, XR, N, 128);

    fill_f32_k<<<blk((long long)N), TB, 0, stream>>>(MAXB, -3.4e38f, N);
    fill_f32_k<<<blk((long long)N), TB, 0, stream>>>(DEN, 0.0f, N);
    fill_f32_k<<<blk((long long)N * FDIM), TB, 0, stream>>>(AGG, 0.0f, N * FDIM);
    edge_logits_max_k<1, 128><<<blk((long long)E), TB, 0, stream>>>(XL, XR, esrc, edst, att3, LOG, MAXB, E);
    edge_accum_k<1, 128><<<blk((long long)E), TB, 0, stream>>>(XL, esrc, edst, LOG, MAXB, DEN, AGG, E);
    finalize_k<1, 128, false><<<blk((long long)N * FDIM), TB, 0, stream>>>(AGG, DEN, bias3, HB, N);

    // ===== Pool + head GEMM [128 x 512] @ [512 x 128] =====
    pool_k<<<N_GRAPH, 128, 0, stream>>>(HB, P16);
    pack_b_k<<<blk(8 * 16 * 512), TB, 0, stream>>>(wh, BPa, 512, 512, 128, 8 * 16 * 512);
    gemm_wmma_k<<<gemm_grid(128, 128), TB, 0, stream>>>(P16, 512, BPa, bh, out, 128, 128);
}
